// SwinUNet_75557064671592
// MI455X (gfx1250) — matrix operand order repaired and checked by reading, not compiled, not run on hardware
//
#include <hip/hip_runtime.h>
#include <hip/hip_bf16.h>
#include <math.h>

// ---------------------------------------------------------------------------
// Swin-UNet forward for MI455X (gfx1250).
// bf16 WMMA (v_wmma_f32_16x16x32_bf16) for all GEMMs, fused bias/GELU/residual
// epilogues. LDS tiles are stored in *fragment-major* order so each wave loads
// a whole 16xbf16 WMMA fragment as one 32B vector (2x ds_load_b128) instead of
// 16 scalar ds_load_u16 + v_mov_b16 packing. Window attention fully in LDS,
// shift/roll folded into index math.
// ---------------------------------------------------------------------------

typedef __bf16 bf16;
typedef __attribute__((ext_vector_type(16))) __bf16    v16bf;
typedef __attribute__((ext_vector_type(8)))  float     v8f;
typedef __attribute__((ext_vector_type(8)))  unsigned  v8u;

#define BATCH 8
#define NEGMASK (-1.0e9f)

// ---------------- device helpers ----------------

__device__ inline unsigned short bfbits(float f) {
  unsigned u = __builtin_bit_cast(unsigned, f);
  u += 0x7FFFu + ((u >> 16) & 1u);                 // round-to-nearest-even
  return (unsigned short)(u >> 16);
}
__device__ inline unsigned pack2(float lo, float hi) {
  return (unsigned)bfbits(lo) | ((unsigned)bfbits(hi) << 16);
}

// A-fragment K index for 16-bit A 16x32 (ISA 7.12.2): element e of lane ->
//   K = (e&7) + 16*(e>>3) + 8*(lane>>4)
// B-fragment K index (same as A): K = (e&7) + 16*(e>>3) + 8*(lane>>4)
// Staging stores dword d = elements (2d, 2d+1) at [tile][lane][d].
__device__ inline int kA_of_d(int d, int lane) {   // K of element e=2d
  int e0 = 2 * d;
  return (e0 & 7) + ((e0 >> 3) << 4) + ((lane >> 4) << 3);
}
__device__ inline int kB_of_d(int d, int lane) {
  int e0 = 2 * d;
  return (e0 & 7) + ((e0 >> 3) << 4) + ((lane >> 4) << 3);
}

__device__ inline v16bf frag_ld(const unsigned* p) {
  v8u u = *reinterpret_cast<const v8u*>(p);        // 32B -> ds_load_b128 x2
  return __builtin_bit_cast(v16bf, u);
}
__device__ inline v8f wmma_bf16(v16bf a, v16bf b, v8f c) {
  return __builtin_amdgcn_wmma_f32_16x16x32_bf16(false, a, false, b,
                                                 (short)0, c, false, false);
}
__device__ inline float gelu_f(float x) {
  float x3 = x * x * x;
  return 0.5f * x * (1.0f + tanhf(0.7978845608028654f * (x + 0.044715f * x3)));
}

// ---------------- GEMM: Out = op(A @ W + bias) [+ resid] ----------------
// A: MxK fp32, W: KxN fp32, tile 128x32, K-step 32, fragment-major bf16 LDS.
// Requires M%128==0, N%32==0, K%32==0 (holds at every call site).
#define OP_NONE 0
#define OP_GELU 1

__global__ __launch_bounds__(256) void k_gemm(
    const float* __restrict__ A, const float* __restrict__ W,
    const float* __restrict__ bias, const float* __restrict__ resid,
    float* __restrict__ Out, long M, int N, int K, int op) {
  __shared__ __align__(16) unsigned Asw[8][32][8];   // 8 row tiles, frag order
  __shared__ __align__(16) unsigned Bsw[2][32][8];   // 2 col tiles, frag order
  const int tid  = threadIdx.x;
  const int lane = tid & 31;
  const int wid  = tid >> 5;                 // 0..7 -> row tile
  const long m0  = (long)blockIdx.x * 128;
  const long n0  = (long)blockIdx.y * 32;

  v8f acc0 = {}, acc1 = {};

  for (int k0 = 0; k0 < K; k0 += 32) {
    if (k0 + 32 < K)                          // -> global_prefetch_b8
      __builtin_prefetch(&W[(long)(k0 + 32) * N + n0 + lane], 0, 1);
#pragma unroll
    for (int it = 0; it < 8; ++it) {          // A tile: 2048 dwords
      int lin = tid + it * 256;
      int d = lin & 7, l = (lin >> 3) & 31, t = lin >> 8;
      int m = t * 16 + (l & 15);
      int k = kA_of_d(d, l);
      float2 av = *reinterpret_cast<const float2*>(&A[(m0 + m) * (long)K + k0 + k]);
      Asw[t][l][d] = pack2(av.x, av.y);
    }
#pragma unroll
    for (int it = 0; it < 2; ++it) {          // W tile: 512 dwords
      int lin = tid + it * 256;
      int l = lin & 31, d = (lin >> 5) & 7, c = lin >> 8;
      int n = c * 16 + (l & 15);
      int kb = kB_of_d(d, l);
      float w0 = W[(long)(k0 + kb) * N + n0 + n];
      float w1 = W[(long)(k0 + kb + 1) * N + n0 + n];
      Bsw[c][l][d] = pack2(w0, w1);
    }
    __syncthreads();
    v16bf af = frag_ld(&Asw[wid][lane][0]);
    v16bf b0 = frag_ld(&Bsw[0][lane][0]);
    v16bf b1 = frag_ld(&Bsw[1][lane][0]);
    acc0 = wmma_bf16(af, b0, acc0);
    acc1 = wmma_bf16(af, b1, acc1);
    __syncthreads();
  }
#pragma unroll
  for (int r = 0; r < 8; ++r) {
    long row = m0 + wid * 16 + r + ((lane >> 4) << 3);   // D layout
    long c0 = n0 + (lane & 15);
    long c1 = c0 + 16;
    float v0 = acc0[r], v1 = acc1[r];
    if (bias) { v0 += bias[c0]; v1 += bias[c1]; }
    if (op == OP_GELU) { v0 = gelu_f(v0); v1 = gelu_f(v1); }
    if (resid) { v0 += resid[row * N + c0]; v1 += resid[row * N + c1]; }
    Out[row * N + c0] = v0;
    Out[row * N + c1] = v1;
  }
}

// ---------------- LayerNorm: one wave per token ----------------
__global__ __launch_bounds__(256) void k_layernorm(
    const float* __restrict__ in, const float* __restrict__ g,
    const float* __restrict__ bb, float* __restrict__ out, int C) {
  int lane = threadIdx.x & 31, wid = threadIdx.x >> 5;
  long tok = (long)blockIdx.x * 8 + wid;
  const float* p = in + tok * C;
  float s = 0.f, ss = 0.f;
  for (int c = lane; c < C; c += 32) { float v = p[c]; s += v; ss += v * v; }
#pragma unroll
  for (int o = 16; o; o >>= 1) { s += __shfl_xor(s, o); ss += __shfl_xor(ss, o); }
  float mean = s / C;
  float rstd = rsqrtf(ss / C - mean * mean + 1e-5f);
  float* q = out + tok * C;
  for (int c = lane; c < C; c += 32)
    q[c] = (p[c] - mean) * rstd * g[c] + bb[c];
}

// ---------------- Window attention (one WG per batch*window*head) ----------
// 64-token window, head dim 32. Shift via gather/scatter index math.
// Q/K/V/P staged in fragment-major LDS so WMMA operands load as b128 pairs.
__global__ __launch_bounds__(128) void k_win_attn(
    const float* __restrict__ qkv, const float* __restrict__ pos,
    float* __restrict__ obuf, int H, int C, int heads, int shifted) {
  const int nw = H >> 3;
  const int nWin = nw * nw;
  int blk  = blockIdx.x;
  int head = blk % heads;
  int win  = (blk / heads) % nWin;
  int b    = blk / (heads * nWin);
  int win_h = win / nw, win_w = win % nw;

  __shared__ __align__(16) unsigned Qw[4][32][8];     // A frags: 4 row tiles
  __shared__ __align__(16) unsigned Kw[4][32][8];     // B frags: 4 col tiles
  __shared__ __align__(16) unsigned Vw[2][2][32][8];  // B frags: [kstep][ctile]
  __shared__ __align__(16) unsigned Pw[2][4][32][8];  // A frags: [kstep][rtile]
  __shared__ float sc[64][65];
  __shared__ int   toks[64];

  int tid = threadIdx.x, lane = tid & 31, wid = tid >> 5;

  if (tid < 64) {                                 // shifted gather indices
    int li = tid >> 3, lj = tid & 7;
    int gr = win_h * 8 + li, gc = win_w * 8 + lj;
    if (shifted) { gr = (gr + 4) & (H - 1); gc = (gc + 4) & (H - 1); }
    toks[tid] = (b * H + gr) * H + gc;
  }
  __syncthreads();

  const int qoff = head * 32;
  const long C3 = 3 * C;
#pragma unroll
  for (int it = 0; it < 8; ++it) {                // Q: 1024 dwords
    int lin = tid + it * 128;
    int d = lin & 7, l = (lin >> 3) & 31, t = lin >> 8;
    int m = t * 16 + (l & 15);
    int k = kA_of_d(d, l);
    float2 qv = *reinterpret_cast<const float2*>(&qkv[(long)toks[m] * C3 + qoff + k]);
    Qw[t][l][d] = pack2(qv.x, qv.y);
  }
#pragma unroll
  for (int it = 0; it < 8; ++it) {                // K^T: 1024 dwords
    int lin = tid + it * 128;
    int d = lin & 7, l = (lin >> 3) & 31, c = lin >> 8;
    int j = c * 16 + (l & 15);                    // token (N index)
    int kb = kB_of_d(d, l);                       // channel (K index)
    float2 kv = *reinterpret_cast<const float2*>(&qkv[(long)toks[j] * C3 + C + qoff + kb]);
    Kw[c][l][d] = pack2(kv.x, kv.y);
  }
#pragma unroll
  for (int it = 0; it < 8; ++it) {                // V: 1024 dwords (pack along K=token)
    int lin = tid + it * 128;
    int d = lin & 7, l = (lin >> 3) & 31, cc = (lin >> 8) & 1, kk = lin >> 9;
    int n = cc * 16 + (l & 15);                   // channel (N index)
    int j0 = kk * 32 + kB_of_d(d, l);             // token (K index)
    float va = qkv[(long)toks[j0]     * C3 + 2 * C + qoff + n];
    float vb = qkv[(long)toks[j0 + 1] * C3 + 2 * C + qoff + n];
    Vw[kk][cc][l][d] = pack2(va, vb);
  }
  __syncthreads();

  {                                               // S = Q @ K^T (64x64, K=32)
    v16bf aq = frag_ld(&Qw[wid][lane][0]);
#pragma unroll
    for (int tc = 0; tc < 4; ++tc) {
      v8f acc = {};
      acc = wmma_bf16(aq, frag_ld(&Kw[tc][lane][0]), acc);
#pragma unroll
      for (int r = 0; r < 8; ++r)
        sc[wid * 16 + r + ((lane >> 4) << 3)][tc * 16 + (lane & 15)] = acc[r];
    }
  }
  __syncthreads();

  if (tid < 64) {                                 // softmax per row
    int i = tid, ir = i >> 3, ic = i & 7;
    bool mUL = shifted && (win_h == nw - 1);
    bool mLR = shifted && (win_w == nw - 1);
    const float scale = 0.17677669529663687f;     // 1/sqrt(32)
    float rowv[64];
    float mx = -1e30f;
#pragma unroll
    for (int j = 0; j < 64; ++j) {
      int jr = j >> 3, jc = j & 7;
      float v = sc[i][j] * scale + pos[(jr - ir + 7) * 15 + (jc - ic + 7)];
      if (mUL && ((ir >= 4) != (jr >= 4))) v += NEGMASK;
      if (mLR && ((ic >= 4) != (jc >= 4))) v += NEGMASK;
      rowv[j] = v;
      mx = fmaxf(mx, v);
    }
    float sum = 0.f;
#pragma unroll
    for (int j = 0; j < 64; ++j) { float e = __expf(rowv[j] - mx); rowv[j] = e; sum += e; }
    float inv = 1.0f / sum;
    // scatter probability pairs straight into P's A-fragment slots
#pragma unroll
    for (int j = 0; j < 64; j += 2) {
      int kk = j >> 5;
      int l2 = (i & 15) + (((j >> 3) & 1) << 4);
      int e0 = (j & 7) + (((j >> 4) & 1) << 3);
      Pw[kk][i >> 4][l2][e0 >> 1] = pack2(rowv[j] * inv, rowv[j + 1] * inv);
    }
  }
  __syncthreads();

#pragma unroll
  for (int tc = 0; tc < 2; ++tc) {                // O = P @ V (64x32, K=64)
    v8f acc = {};
#pragma unroll
    for (int kk = 0; kk < 2; ++kk) {
      v16bf af  = frag_ld(&Pw[kk][wid][lane][0]);
      v16bf bfv = frag_ld(&Vw[kk][tc][lane][0]);
      acc = wmma_bf16(af, bfv, acc);
    }
#pragma unroll
    for (int r = 0; r < 8; ++r) {
      int t = wid * 16 + r + ((lane >> 4) << 3);
      obuf[(long)toks[t] * C + qoff + tc * 16 + (lane & 15)] = acc[r];
    }
  }
}

// ---------------- index-shuffle kernels ----------------

__global__ void k_patch_embed(const float* __restrict__ x,
                              const float* __restrict__ w,
                              const float* __restrict__ bvec,
                              float* __restrict__ out, long total) {
  long idx = (long)blockIdx.x * 256 + threadIdx.x;
  if (idx >= total) return;
  int oc = idx % 96;
  long tok = idx / 96;
  int b = tok / 4096, rem = tok % 4096;
  int r = rem >> 6, cg = rem & 63;
  float acc = bvec[oc];
#pragma unroll
  for (int f = 0; f < 48; ++f) {
    int ch = f >> 4, pr = (f >> 2) & 3, pc = f & 3;
    float xv = x[((long)(b * 3 + ch) * 256 + r * 4 + pr) * 256 + cg * 4 + pc];
    acc += xv * w[f * 96 + oc];
  }
  out[idx] = acc;
}

__global__ void k_merge_gather(const float* __restrict__ in, float* __restrict__ out,
                               int H, int C, long total) {
  long idx = (long)blockIdx.x * 256 + threadIdx.x;
  if (idx >= total) return;
  int C4 = 4 * C, Ho = H >> 1;
  long toko = idx / C4;
  int c = idx % C4, part = c / C, cc = c % C;
  int b = toko / (Ho * Ho), rem = toko % (Ho * Ho);
  int i = rem / Ho, j = rem % Ho;
  int di = (part == 1 || part == 3) ? 1 : 0;     // x0,x1,x2,x3 order
  int dj = (part >= 2) ? 1 : 0;
  out[idx] = in[(((long)b * H + 2 * i + di) * H + 2 * j + dj) * C + cc];
}

__global__ void k_pixel_shuffle(const float* __restrict__ in, float* __restrict__ out,
                                int H, int F, int c2, long total) {
  long idx = (long)blockIdx.x * 256 + threadIdx.x;
  if (idx >= total) return;
  int Ho = H * F;
  long toko = idx / c2;
  int c = idx % c2;
  int b = toko / ((long)Ho * Ho), rem = toko % ((long)Ho * Ho);
  int r = rem / Ho, col = rem % Ho;
  int i = r / F, p = r % F, j = col / F, q = col % F;
  out[idx] = in[(((long)b * H + i) * H + j) * (F * F * c2) + (p * F + q) * c2 + c];
}

__global__ void k_concat(const float* __restrict__ a, const float* __restrict__ bsrc,
                         float* __restrict__ out, int Ca, int Cb, long total) {
  long idx = (long)blockIdx.x * 256 + threadIdx.x;
  if (idx >= total) return;
  int Cs = Ca + Cb;
  long t = idx / Cs;
  int c = idx % Cs;
  out[idx] = (c < Ca) ? a[t * Ca + c] : bsrc[t * Cb + (c - Ca)];
}

__global__ void k_head(const float* __restrict__ y, const float* __restrict__ hw,
                       const float* __restrict__ hb, float* __restrict__ out, long M) {
  long t = (long)blockIdx.x * 256 + threadIdx.x;
  if (t >= M) return;
  float acc = hb[0];
#pragma unroll
  for (int c = 0; c < 96; ++c) acc += y[t * 96 + c] * hw[c];
  out[t] = acc;
}

// ---------------- host orchestration ----------------

struct BlockP {
  const float *bout, *pos, *wout, *wqkv, *b1, *b2;
  const float *ln1b, *ln1g, *ln2b, *ln2g, *w1, *w2;
};
struct DecStage { BlockP blk[2]; const float *fuse_b, *fuse_w, *up_exp, *up_nb, *up_ng; bool has_up; };
struct EncStage { BlockP blk[2]; const float *m_nb, *m_ng, *m_red; bool has_merge; };

static inline void gemm(hipStream_t st, const float* A, const float* W, const float* bias,
                        const float* resid, float* out, long M, int N, int K, int op) {
  dim3 g((unsigned)(M / 128), (unsigned)(N / 32));
  k_gemm<<<g, 256, 0, st>>>(A, W, bias, resid, out, M, N, K, op);
}
static inline void lnorm(hipStream_t st, const float* in, const float* g, const float* b,
                         float* out, long M, int C) {
  k_layernorm<<<(unsigned)(M / 8), 256, 0, st>>>(in, g, b, out, C);
}
static inline unsigned nblk(long total) { return (unsigned)((total + 255) / 256); }

static void run_block(hipStream_t st, float*& cur, float*& alt, float* t1, float* qkvb,
                      float* abuf, float* hbuf, const BlockP& p,
                      long tokens, int C, int heads, int H, int shifted) {
  lnorm(st, cur, p.ln1g, p.ln1b, t1, tokens, C);
  gemm(st, t1, p.wqkv, nullptr, nullptr, qkvb, tokens, 3 * C, C, OP_NONE);
  int nWin = (H >> 3) * (H >> 3);
  k_win_attn<<<(unsigned)(BATCH * nWin * heads), 128, 0, st>>>(qkvb, p.pos, abuf, H, C, heads, shifted);
  gemm(st, abuf, p.wout, p.bout, cur, alt, tokens, C, C, OP_NONE);
  { float* t = cur; cur = alt; alt = t; }
  lnorm(st, cur, p.ln2g, p.ln2b, t1, tokens, C);
  gemm(st, t1, p.w1, p.b1, nullptr, hbuf, tokens, 4 * C, C, OP_GELU);
  gemm(st, hbuf, p.w2, p.b2, cur, alt, tokens, C, 4 * C, OP_NONE);
  { float* t = cur; cur = alt; alt = t; }
}

extern "C" void kernel_launch(void* const* d_in, const int* in_sizes, int n_in,
                              void* d_out, int out_size, void* d_ws, size_t ws_size,
                              hipStream_t stream) {
  (void)in_sizes; (void)n_in; (void)out_size; (void)ws_size;
  static const int HEADS[4] = {3, 6, 12, 24};

  // ---- pytree leaf order: x, then params (dict keys alphabetized) ----
  const float* x = (const float*)d_in[0];
  int ip = 1;
  auto nextp = [&]() -> const float* { return (const float*)d_in[ip++]; };
  auto read_block = [&](BlockP& b) {
    b.bout = nextp(); b.pos = nextp(); b.wout = nextp(); b.wqkv = nextp();
    b.b1 = nextp();  b.b2 = nextp();
    b.ln1b = nextp(); b.ln1g = nextp(); b.ln2b = nextp(); b.ln2g = nextp();
    b.w1 = nextp();  b.w2 = nextp();
  };
  DecStage dec[4];
  for (int s = 0; s < 4; ++s) {                  // dec list: i = 3,2,1,0
    int ist = 3 - s;
    read_block(dec[s].blk[0]); read_block(dec[s].blk[1]);
    dec[s].has_up = (ist < 3);
    if (ist < 3) {
      dec[s].fuse_b = nextp(); dec[s].fuse_w = nextp();
      dec[s].up_exp = nextp(); dec[s].up_nb = nextp(); dec[s].up_ng = nextp();
    }
  }
  const float* emb_b  = nextp(); const float* emb_nb = nextp();
  const float* emb_ng = nextp(); const float* emb_w  = nextp();
  EncStage enc[4];
  for (int i = 0; i < 4; ++i) {
    read_block(enc[i].blk[0]); read_block(enc[i].blk[1]);
    enc[i].has_merge = (i < 3);
    if (i < 3) { enc[i].m_nb = nextp(); enc[i].m_ng = nextp(); enc[i].m_red = nextp(); }
  }
  const float* fin_exp = nextp(); const float* fin_nb = nextp(); const float* fin_ng = nextp();
  const float* head_b = nextp(); const float* head_w = nextp();

  // ---- workspace arena (floats) ----
  float* wsf = (float*)d_ws;
  size_t off = 0;
  auto arena = [&](size_t n) { float* p = wsf + off; off += n; return p; };
  float* cur  = arena(3145728);
  float* alt  = arena(3145728);
  float* t1   = arena(3145728);
  float* qkvb = arena(9437184);
  float* abuf = arena(3145728);
  float* hbuf = arena(12582912);
  float* skip[4] = { arena(3145728), arena(1572864), arena(786432), arena(393216) };
  float* fexp = arena(50331648);
  float* fy   = arena(50331648);

  // ---- patch embed ----
  long tok0 = (long)BATCH * 64 * 64;
  k_patch_embed<<<nblk(tok0 * 96), 256, 0, stream>>>(x, emb_w, emb_b, t1, tok0 * 96);
  lnorm(stream, t1, emb_ng, emb_nb, cur, tok0, 96);

  // ---- encoder ----
  for (int i = 0; i < 4; ++i) {
    int C = 96 << i, H = 64 >> i, heads = HEADS[i];
    long tokens = (long)BATCH * H * H;
    run_block(stream, cur, alt, t1, qkvb, abuf, hbuf, enc[i].blk[0], tokens, C, heads, H, 0);
    run_block(stream, cur, alt, t1, qkvb, abuf, hbuf, enc[i].blk[1], tokens, C, heads, H, 1);
    hipMemcpyAsync(skip[i], cur, tokens * C * sizeof(float), hipMemcpyDeviceToDevice, stream);
    if (enc[i].has_merge) {
      long total = tokens * C;                   // (tokens/4) x 4C
      k_merge_gather<<<nblk(total), 256, 0, stream>>>(cur, hbuf, H, C, total);
      lnorm(stream, hbuf, enc[i].m_ng, enc[i].m_nb, qkvb, tokens / 4, 4 * C);
      gemm(stream, qkvb, enc[i].m_red, nullptr, nullptr, alt, tokens / 4, 2 * C, 4 * C, OP_NONE);
      { float* t = cur; cur = alt; alt = t; }
    }
  }

  // ---- decoder ----
  for (int s = 0; s < 4; ++s) {
    int ist = 3 - s;
    int C = 96 << ist, H = 64 >> ist, heads = HEADS[ist];
    long tokens = (long)BATCH * H * H;
    if (dec[s].has_up) {
      gemm(stream, cur, dec[s].up_exp, nullptr, nullptr, hbuf, tokens / 4, 4 * C, 2 * C, OP_NONE);
      long total = tokens * C;
      k_pixel_shuffle<<<nblk(total), 256, 0, stream>>>(hbuf, abuf, H / 2, 2, C, total);
      lnorm(stream, abuf, dec[s].up_ng, dec[s].up_nb, t1, tokens, C);
      long ctot = tokens * 2 * C;
      k_concat<<<nblk(ctot), 256, 0, stream>>>(t1, skip[ist], hbuf, C, C, ctot);
      gemm(stream, hbuf, dec[s].fuse_w, dec[s].fuse_b, nullptr, alt, tokens, C, 2 * C, OP_NONE);
      { float* t = cur; cur = alt; alt = t; }
    }
    run_block(stream, cur, alt, t1, qkvb, abuf, hbuf, dec[s].blk[0], tokens, C, heads, H, 0);
    run_block(stream, cur, alt, t1, qkvb, abuf, hbuf, dec[s].blk[1], tokens, C, heads, H, 1);
  }

  // ---- final expand (F=4) + LN + head ----
  gemm(stream, cur, fin_exp, nullptr, nullptr, fexp, tok0, 1536, 96, OP_NONE);
  long ftot = (long)BATCH * 256 * 256 * 96;
  k_pixel_shuffle<<<nblk(ftot), 256, 0, stream>>>(fexp, fy, 64, 4, 96, ftot);
  long ftok = (long)BATCH * 256 * 256;
  lnorm(stream, fy, fin_ng, fin_nb, fexp, ftok, 96);
  k_head<<<nblk(ftok), 256, 0, stream>>>(fexp, head_w, head_b, (float*)d_out, ftok);
}
